// TreeEncoder_13005160972560
// MI455X (gfx1250) — matrix operand order repaired and checked by reading, not compiled, not run on hardware
//
#include <hip/hip_runtime.h>
#include <hip/hip_bf16.h>

// Problem constants (from reference): B=64, S=4096, D=256, A=8, L=4, P=512
#define B_    64
#define S_    4096
#define D_    256
#define A_    8
#define NI_   512          // number of internal nodes ((S-2)//A + 1)
#define P_    512
#define TM    32           // (b,parent) rows per block = two 16-row WMMA M tiles
#define NEGV  (-1000000000.0f)

typedef __attribute__((ext_vector_type(16))) __bf16 v16bf;
typedef __attribute__((ext_vector_type(8)))  float  v8f;

union Frag16 { uint4 u[2]; v16bf v; };

__device__ __forceinline__ float fast_tanh(float v) {
#if __has_builtin(__builtin_amdgcn_tanhf)
  return __builtin_amdgcn_tanhf(v);     // gfx1250 native v_tanh_f32
#else
  return tanhf(v);
#endif
}

// ---------------------------------------------------------------------------
// Convert W (f32, [2D=512][D=256] row-major) to transposed bf16 Wt[n][k],
// so a WMMA B-fragment lane can read 8 consecutive K values (16B) per load.
// ---------------------------------------------------------------------------
__global__ void prep_w_kernel(const float* __restrict__ W, __bf16* __restrict__ Wt) {
  int idx = blockIdx.x * blockDim.x + threadIdx.x;   // 512*256 threads
  int k  = idx >> 8;                                  // 0..511
  int nn = idx & 255;                                 // 0..255
  Wt[(size_t)nn * 512 + k] = (__bf16)W[idx];
}

// ---------------------------------------------------------------------------
// One tree level. Each block: TM=32 (b,parent) rows, full N=256 output.
//  Phase 1 (4 rows per wave): q/children loads, dots via shfl_xor butterfly,
//           softmax, mix -> comb=[mix,q] in LDS as bf16 (32x512).
//  Phase 2: 8 waves x (2 M-tiles x 2 N-tiles), K-loop 512 in steps of 32,
//           bf16 WMMA with f32 accumulators (B frags reused across M tiles);
//           bias + native tanh; scatter to upd[b, p, :].
// ---------------------------------------------------------------------------
__global__ __launch_bounds__(256)
void tree_level_kernel(const float* __restrict__ x,
                       const int*   __restrict__ parent,    // this level's row, P_ ints
                       const int*   __restrict__ children,  // this level's rows, P_*A_ ints
                       const float* __restrict__ bias,      // D_ floats
                       const __bf16* __restrict__ Wt,       // [256][512] bf16 (transposed W)
                       float*       __restrict__ upd,       // [B_][NI_][D_] f32
                       int n, int total) {
  __shared__ __align__(16) __bf16 comb[TM][520];   // 512 + pad, rows 16B aligned
  __shared__ int rowP[TM];
  __shared__ int rowB[TM];

  const int tid  = threadIdx.x;
  const int wave = tid >> 5;
  const int lane = tid & 31;
  const int blockBase = blockIdx.x * TM;

  // ------------------------- Phase 1: build comb rows -------------------------
  for (int rr = 0; rr < 4; ++rr) {
    const int r   = wave + rr * 8;           // row within tile, 0..31
    const int gid = blockBase + r;           // global (b,parent) pair id
    int b = 0, i = 0, p = -1;
    if (gid < total) { b = gid / n; i = gid - b * n; p = parent[i]; }
    if (lane == 0) { rowP[r] = p; rowB[r] = b; }

    const int dbase = lane * 8;              // this lane owns dims dbase..dbase+7

    if (p >= 0) {
      float q[8];
      float ctx[8][8];
      float dotp[8];
      int   cc[8];

      const float* qs = x + ((size_t)b * S_ + p) * D_ + dbase;
      #pragma unroll
      for (int d = 0; d < 8; ++d) q[d] = qs[d];

      #pragma unroll
      for (int a = 0; a < A_; ++a) {
        const int c = children[i * A_ + a];
        cc[a] = c;
        float s = 0.0f;
        if (c >= 0) {
          // internal children (<NI_) were produced by the deeper level
          const float* cs = (c < NI_)
              ? (upd + ((size_t)b * NI_ + c) * D_ + dbase)
              : (x   + ((size_t)b * S_  + c) * D_ + dbase);
          #pragma unroll
          for (int d = 0; d < 8; ++d) { ctx[a][d] = cs[d]; s += q[d] * cs[d]; }
        } else {
          #pragma unroll
          for (int d = 0; d < 8; ++d) ctx[a][d] = 0.0f;   // sentinel row is zeros
        }
        dotp[a] = s;
      }

      // butterfly reduce the 8 partial dots across the wave (all lanes get sums)
      #pragma unroll
      for (int off = 16; off > 0; off >>= 1) {
        #pragma unroll
        for (int a = 0; a < A_; ++a) dotp[a] += __shfl_xor(dotp[a], off, 32);
      }

      // softmax over the 8 children with mask (score = NEG for invalid)
      float mx = -3.4e38f;
      #pragma unroll
      for (int a = 0; a < A_; ++a) {
        float sc = (cc[a] >= 0) ? dotp[a] : NEGV;
        mx = fmaxf(mx, sc);
      }
      float e[8]; float esum = 0.0f;
      #pragma unroll
      for (int a = 0; a < A_; ++a) {
        e[a] = (cc[a] >= 0) ? __expf(dotp[a] - mx) : 0.0f;
        esum += e[a];
      }
      const float inv = 1.0f / esum;

      // mix and write comb row: [mix (0..255) | q (256..511)] as bf16
      #pragma unroll
      for (int d = 0; d < 8; ++d) {
        float m = 0.0f;
        #pragma unroll
        for (int a = 0; a < A_; ++a) m += e[a] * ctx[a][d];
        m *= inv;
        comb[r][dbase + d]        = (__bf16)m;
        comb[r][256 + dbase + d]  = (__bf16)q[d];
      }
    } else {
      // padded row: zero so the WMMA tile stays clean
      #pragma unroll
      for (int d = 0; d < 8; ++d) {
        comb[r][dbase + d]       = (__bf16)0.0f;
        comb[r][256 + dbase + d] = (__bf16)0.0f;
      }
    }
  }

  __syncthreads();

  // ------------------------- Phase 2: WMMA GEMM -------------------------
  // Wave w owns output columns [32w, 32w+32): two 16-wide N tiles,
  // and both 16-row M tiles (rows 0..15 and 16..31) reuse the B fragments.
  const int col0  = wave * 32;
  const int colL  = lane & 15;
  const int rowA  = lane & 15;                 // A-matrix row held by this lane
  const int koffA = (lane < 16) ? 0 : 8;       // A lane K groups: {0..7,16..23} / {8..15,24..31}
  const int kbB   = (lane < 16) ? 0 : 8;       // B lane K groups: {0..7,16..23} / {8..15,24..31}

  v8f acc00 = {};   // M tile 0, N tile 0
  v8f acc01 = {};   // M tile 0, N tile 1
  v8f acc10 = {};   // M tile 1, N tile 0
  v8f acc11 = {};   // M tile 1, N tile 1

  #pragma unroll 2
  for (int k0 = 0; k0 < 512; k0 += 32) {
    Frag16 af0, af1;
    af0.u[0] = *(const uint4*)&comb[rowA][k0 + koffA];             // K = k0+koffA   .. +7
    af0.u[1] = *(const uint4*)&comb[rowA][k0 + koffA + 16];        // K = k0+koffA+16.. +23
    af1.u[0] = *(const uint4*)&comb[rowA + 16][k0 + koffA];
    af1.u[1] = *(const uint4*)&comb[rowA + 16][k0 + koffA + 16];

    Frag16 bf0, bf1;
    const uint4* w0 = (const uint4*)(Wt + (size_t)(col0 + colL)      * 512 + k0 + kbB);
    const uint4* w1 = (const uint4*)(Wt + (size_t)(col0 + 16 + colL) * 512 + k0 + kbB);
    bf0.u[0] = w0[0]; bf0.u[1] = w0[2];                            // K = k0+kbB..+7, k0+kbB+16..+23
    bf1.u[0] = w1[0]; bf1.u[1] = w1[2];

    acc00 = __builtin_amdgcn_wmma_f32_16x16x32_bf16(false, af0.v, false, bf0.v,
                                                    (short)0, acc00, false, false);
    acc01 = __builtin_amdgcn_wmma_f32_16x16x32_bf16(false, af0.v, false, bf1.v,
                                                    (short)0, acc01, false, false);
    acc10 = __builtin_amdgcn_wmma_f32_16x16x32_bf16(false, af1.v, false, bf0.v,
                                                    (short)0, acc10, false, false);
    acc11 = __builtin_amdgcn_wmma_f32_16x16x32_bf16(false, af1.v, false, bf1.v,
                                                    (short)0, acc11, false, false);
  }

  // bias + tanh + scatter. C/D layout: VGPR v -> M = v + (lane<16 ? 0 : 8), N = lane&15.
  const float bias0 = bias[col0 + colL];
  const float bias1 = bias[col0 + 16 + colL];
  const int mbase = (lane < 16) ? 0 : 8;

  #pragma unroll
  for (int v = 0; v < 8; ++v) {
    const int m0 = v + mbase;          // row in M tile 0
    const int m1 = m0 + 16;            // row in M tile 1
    const int gid0 = blockBase + m0;
    const int gid1 = blockBase + m1;
    if (gid0 < total) {
      float* dst = upd + ((size_t)rowB[m0] * NI_ + rowP[m0]) * D_;
      dst[col0 + colL]      = fast_tanh(acc00[v] + bias0);
      dst[col0 + 16 + colL] = fast_tanh(acc01[v] + bias1);
    }
    if (gid1 < total) {
      float* dst = upd + ((size_t)rowB[m1] * NI_ + rowP[m1]) * D_;
      dst[col0 + colL]      = fast_tanh(acc10[v] + bias0);
      dst[col0 + 16 + colL] = fast_tanh(acc11[v] + bias1);
    }
  }
}

// ---------------------------------------------------------------------------
// Emit root node (internal node 0) as the output: out[b][d] = upd[b][0][d]
// ---------------------------------------------------------------------------
__global__ void copy_root_kernel(const float* __restrict__ upd, float* __restrict__ out) {
  int idx = blockIdx.x * blockDim.x + threadIdx.x;   // B_*D_ threads
  int b = idx >> 8;
  int d = idx & 255;
  out[idx] = upd[((size_t)b * NI_) * D_ + d];
}

// ---------------------------------------------------------------------------
extern "C" void kernel_launch(void* const* d_in, const int* in_sizes, int n_in,
                              void* d_out, int out_size, void* d_ws, size_t ws_size,
                              hipStream_t stream) {
  (void)in_sizes; (void)n_in; (void)out_size; (void)ws_size;

  const float* x        = (const float*)d_in[0];   // [B][S][D] f32
  const float* W        = (const float*)d_in[1];   // [2D][D]  f32
  const float* bias     = (const float*)d_in[2];   // [D]      f32
  const int*   parent   = (const int*)  d_in[3];   // [L][P]   i32
  const int*   children = (const int*)  d_in[4];   // [L][P][A] i32

  // Workspace layout: [0, 256KB) = Wt bf16 [256][512]; then upd f32 [B][NI][D] (32 MB)
  __bf16* Wt  = (__bf16*)d_ws;
  float*  upd = (float*)((char*)d_ws + (size_t)512 * 256 * sizeof(__bf16));

  prep_w_kernel<<<512, 256, 0, stream>>>(W, Wt);

  // Static per-level internal node counts: depth0={0}, depth1={1..8},
  // depth2={9..72}, depth3={73..511}
  const int ns[4] = {1, 8, 64, 439};
  for (int lvl = 3; lvl >= 0; --lvl) {
    const int n      = ns[lvl];
    const int total  = B_ * n;
    const int blocks = (total + TM - 1) / TM;
    tree_level_kernel<<<blocks, 256, 0, stream>>>(
        x, parent + lvl * P_, children + lvl * P_ * A_,
        bias, Wt, upd, n, total);
  }

  copy_root_kernel<<<B_, 256, 0, stream>>>(upd, (float*)d_out);
}
